// EGNNLayer_5841155523150
// MI455X (gfx1250) — matrix operand order repaired and checked by reading, not compiled, not run on hardware
//
#include <hip/hip_runtime.h>
#include <hip/hip_bf16.h>

typedef __attribute__((ext_vector_type(16))) _Float16 v16h;
typedef __attribute__((ext_vector_type(8)))  _Float16 v8h;
typedef __attribute__((ext_vector_type(8)))  float    v8f;

#define BB 4
#define NN 512
#define DS 128
#define ED 32
#define MAXD2 22500.0f   // 150^2

static __device__ __forceinline__ v8f wmma16(v16h a, v16h b, v8f c) {
    return __builtin_amdgcn_wmma_f32_16x16x32_f16(false, a, false, b, (short)0, c, false, false);
}

// fast silu: x * rcp(1+exp(-x))  (v_exp_f32 + v_rcp_f32, no IEEE divide)
static __device__ __forceinline__ float silu(float v) {
    return v * __builtin_amdgcn_rcpf(1.0f + __expf(-v));
}

// ---------------------------------------------------------------------------
// Kernel 1: per-node precompute  U = s @ ew1[0:128] + eb1,  V = s @ ew1[128:256]
// grid = B*N blocks, 64 threads
// ---------------------------------------------------------------------------
__global__ __launch_bounds__(64) void uv_kernel(const float* __restrict__ s,
                                                const float* __restrict__ ew1,
                                                const float* __restrict__ eb1,
                                                float* __restrict__ U,
                                                float* __restrict__ V) {
    const int node = blockIdx.x;
    const int t = threadIdx.x;
    __shared__ float srow[DS];
    srow[t]      = s[node * DS + t];
    srow[t + 64] = s[node * DS + t + 64];
    __syncthreads();
    const int k = t & 31;
    const bool isV = t >= 32;
    const float* w = ew1 + (isV ? DS * ED : 0);
    float acc = isV ? 0.0f : eb1[k];
#pragma unroll 8
    for (int r = 0; r < DS; ++r) acc += srow[r] * w[r * ED + k];
    (isV ? V : U)[node * ED + k] = acc;
}

// ---------------------------------------------------------------------------
// Kernel 2: fused edge MLP + attention + coord MLP + online softmax aggregation
// One wave32 per (b,i). Block = 128 threads = 4 waves. grid = B*N/4.
// ---------------------------------------------------------------------------
__global__ __launch_bounds__(128) void edge_kernel(
    const float* __restrict__ x, const int* __restrict__ mask,
    const float* __restrict__ ew1,
    const float* __restrict__ ew2, const float* __restrict__ eb2,
    const float* __restrict__ aw1, const float* __restrict__ ab1,
    const float* __restrict__ aw2, const float* __restrict__ ab2,
    const float* __restrict__ cw1, const float* __restrict__ cb1,
    const float* __restrict__ cw2, const float* __restrict__ cb2,
    const float* __restrict__ U, const float* __restrict__ V,
    float* __restrict__ MS, float* __restrict__ outX) {

    const int lane = threadIdx.x & 31;
    const int wid  = threadIdx.x >> 5;
    const int row  = blockIdx.x * 4 + wid;   // flat (b*N + i)
    const int b    = row / NN;
    const int col  = lane & 15;              // C-layout column / A-layout row
    const int half = lane >> 4;
    const int kbase = 8 * half;              // A-layout K base for this lane

    __shared__ __align__(16) float4   xt[4][16];       // per-wave x tile (w = mask)
    __shared__ __align__(16) _Float16 ef[4][16][ED];   // per-wave edge_feat restage

    // ---- B fragments: lane%16 = N column of 32x32 weight, element = K row (A-layout K order) ----
    v16h b_ef0, b_ef1, b_a0, b_a1, b_c0, b_c1;
#pragma unroll
    for (int n = 0; n < 16; ++n) {
        const int kk = kbase + n + ((n >= 8) ? 8 : 0);
        b_ef0[n] = (_Float16)ew2[kk * ED + col];
        b_ef1[n] = (_Float16)ew2[kk * ED + 16 + col];
        b_a0[n]  = (_Float16)aw1[kk * ED + col];
        b_a1[n]  = (_Float16)aw1[kk * ED + 16 + col];
        b_c0[n]  = (_Float16)cw1[kk * ED + col];
        b_c1[n]  = (_Float16)cw1[kk * ED + 16 + col];
    }

    // ---- per-lane constants in A-layout K order ----
    float uk[16], wd[16];
#pragma unroll
    for (int e = 0; e < 16; ++e) {
        const int k = kbase + e + ((e >= 8) ? 8 : 0);
        uk[e] = U[row * ED + k];
        wd[e] = ew1[256 * ED + k];
    }
    const float xi0 = x[row * 3 + 0], xi1 = x[row * 3 + 1], xi2 = x[row * 3 + 2];
    const float eb2lo = eb2[col],      eb2hi = eb2[16 + col];
    const float ab1lo = ab1[col],      ab1hi = ab1[16 + col];
    const float cb1lo = cb1[col],      cb1hi = cb1[16 + col];
    const float aw2lo = aw2[col],      aw2hi = aw2[16 + col];
    const float cw2lo = cw2[col],      cw2hi = cw2[16 + col];
    const float ab2s = ab2[0], cb2s = cb2[0];
    const int   maski = mask[row];

    float Mx = -INFINITY, denom = 0.0f;
    float ams_lo = 0.0f, ams_hi = 0.0f;
    float amx0 = 0.0f, amx1 = 0.0f, amx2 = 0.0f;

    for (int j0 = 0; j0 < NN; j0 += 16) {
        __syncthreads();                       // previous-iter LDS reads done
        if (lane < 16) {
            const int j = j0 + lane;
            float4 xm;
            xm.x = x[(b * NN + j) * 3 + 0];
            xm.y = x[(b * NN + j) * 3 + 1];
            xm.z = x[(b * NN + j) * 3 + 2];
            xm.w = (float)mask[b * NN + j];
            xt[wid][lane] = xm;
        }
        __syncthreads();

        if (j0 + 16 < NN)   // warm next V tile
            __builtin_prefetch(V + (b * NN + j0 + 16 + col) * ED, 0, 1);

        // ---- build A fragment of silu(U_i + V_j + d2*w_d) (row = col) ----
        const int jA = j0 + col;
        const float4 xja = xt[wid][col];
        const float dxa = xi0 - xja.x;
        const float dya = xi1 - xja.y;
        const float dza = xi2 - xja.z;
        const float d2a = dxa * dxa + dya * dya + dza * dza;
        const float* vrow = V + (b * NN + jA) * ED + kbase;
        v16h a;
#pragma unroll
        for (int e = 0; e < 8; ++e) {
            const float pre = uk[e] + vrow[e] + d2a * wd[e];
            a[e] = (_Float16)silu(pre);
        }
#pragma unroll
        for (int e = 8; e < 16; ++e) {
            const float pre = uk[e] + vrow[e + 8] + d2a * wd[e];
            a[e] = (_Float16)silu(pre);
        }

        // ---- edge_feat = h @ ew2 + eb2  (two WMMAs for 32 output cols) ----
        v8f c0 = {}, c1 = {};
        c0 = wmma16(a, b_ef0, c0);
        c1 = wmma16(a, b_ef1, c1);
        float EF0[8], EF1[8];
#pragma unroll
        for (int r = 0; r < 8; ++r) { EF0[r] = c0[r] + eb2lo; EF1[r] = c1[r] + eb2hi; }

        // ---- restage edge_feat (C layout -> A layout) through LDS ----
#pragma unroll
        for (int r = 0; r < 8; ++r) {
            const int rr = r + 8 * half;
            ef[wid][rr][col]      = (_Float16)EF0[r];
            ef[wid][rr][16 + col] = (_Float16)EF1[r];
        }
        __syncthreads();
        // A-fragment read: two 16B-aligned K-contiguous runs -> 2x ds_load_b128
        const v8h lo = *reinterpret_cast<const v8h*>(&ef[wid][col][kbase]);
        const v8h hi = *reinterpret_cast<const v8h*>(&ef[wid][col][kbase + 16]);
        const v16h ae = __builtin_shufflevector(lo, hi,
            0, 1, 2, 3, 4, 5, 6, 7, 8, 9, 10, 11, 12, 13, 14, 15);

        // ---- attention & coord hidden layers (four WMMAs) ----
        v8f ah0 = {}, ah1 = {}, ch0 = {}, ch1 = {};
        ah0 = wmma16(ae, b_a0, ah0);
        ah1 = wmma16(ae, b_a1, ah1);
        ch0 = wmma16(ae, b_c0, ch0);
        ch1 = wmma16(ae, b_c1, ch1);

        // ---- 32->1 output layers: reduce over columns (16 lanes / half) ----
        float lg[8], cwv[8];
#pragma unroll
        for (int r = 0; r < 8; ++r) {
            float v = silu(ah0[r] + ab1lo) * aw2lo + silu(ah1[r] + ab1hi) * aw2hi;
            v += __shfl_xor(v, 1, 32); v += __shfl_xor(v, 2, 32);
            v += __shfl_xor(v, 4, 32); v += __shfl_xor(v, 8, 32);
            lg[r] = v + ab2s;
            float w = silu(ch0[r] + cb1lo) * cw2lo + silu(ch1[r] + cb1hi) * cw2hi;
            w += __shfl_xor(w, 1, 32); w += __shfl_xor(w, 2, 32);
            w += __shfl_xor(w, 4, 32); w += __shfl_xor(w, 8, 32);
            cwv[r] = w + cb2s;
        }

        // ---- validity + online softmax update ----
        float dxr[8], dyr[8], dzr[8];
#pragma unroll
        for (int r = 0; r < 8; ++r) {
            const int jr = r + 8 * half;
            const float4 xj = xt[wid][jr];
            dxr[r] = xi0 - xj.x;
            dyr[r] = xi1 - xj.y;
            dzr[r] = xi2 - xj.z;
            const float d2 = dxr[r] * dxr[r] + dyr[r] * dyr[r] + dzr[r] * dzr[r];
            const bool valid = (d2 <= MAXD2) && (maski != 0) && (xj.w != 0.0f);
            if (!valid) lg[r] = -INFINITY;
        }
        float tm = lg[0];
#pragma unroll
        for (int r = 1; r < 8; ++r) tm = fmaxf(tm, lg[r]);
        tm = fmaxf(tm, __shfl_xor(tm, 16, 32));
        const float newM = fmaxf(Mx, tm);
        const float scale = (Mx == -INFINITY) ? 0.0f : __expf(Mx - newM);

        float p[8], ts = 0.0f, pml = 0.0f, pmh = 0.0f;
        float px = 0.0f, py = 0.0f, pz = 0.0f;
#pragma unroll
        for (int r = 0; r < 8; ++r) {
            p[r] = (lg[r] == -INFINITY) ? 0.0f : __expf(lg[r] - newM);
            ts  += p[r];
            pml += p[r] * EF0[r];
            pmh += p[r] * EF1[r];
            const float pw = p[r] * cwv[r];
            px += pw * dxr[r]; py += pw * dyr[r]; pz += pw * dzr[r];
        }
        ts  += __shfl_xor(ts, 16, 32);
        pml += __shfl_xor(pml, 16, 32);
        pmh += __shfl_xor(pmh, 16, 32);
        px  += __shfl_xor(px, 16, 32);
        py  += __shfl_xor(py, 16, 32);
        pz  += __shfl_xor(pz, 16, 32);

        denom  = denom  * scale + ts;
        ams_lo = ams_lo * scale + pml;
        ams_hi = ams_hi * scale + pmh;
        amx0   = amx0   * scale + px;
        amx1   = amx1   * scale + py;
        amx2   = amx2   * scale + pz;
        Mx = newM;
    }

    const float inv = __builtin_amdgcn_rcpf(denom);
    if (half == 0) {
        MS[row * ED + col]      = ams_lo * inv;
        MS[row * ED + 16 + col] = ams_hi * inv;
    }
    if (lane == 0) {
        outX[row * 3 + 0] = xi0 + amx0 * inv;
        outX[row * 3 + 1] = xi1 + amx1 * inv;
        outX[row * 3 + 2] = xi2 + amx2 * inv;
    }
}

// ---------------------------------------------------------------------------
// Kernel 3: node update  s_out = s + MLP([s, m_s])   (160 -> 128 -> 128)
// grid = B*N blocks, 128 threads
// ---------------------------------------------------------------------------
__global__ __launch_bounds__(128) void node_kernel(const float* __restrict__ s,
                                                   const float* __restrict__ MS,
                                                   const float* __restrict__ sw1,
                                                   const float* __restrict__ sb1,
                                                   const float* __restrict__ sw2,
                                                   const float* __restrict__ sb2,
                                                   float* __restrict__ outS) {
    const int node = blockIdx.x;
    const int t = threadIdx.x;
    __shared__ float in[DS + ED];
    __shared__ float hid[DS];
    in[t] = s[node * DS + t];
    if (t < ED) in[DS + t] = MS[node * ED + t];
    __syncthreads();
    float acc = sb1[t];
#pragma unroll 8
    for (int r = 0; r < DS + ED; ++r) acc += in[r] * sw1[r * DS + t];
    hid[t] = silu(acc);
    __syncthreads();
    float acc2 = sb2[t];
#pragma unroll 8
    for (int r = 0; r < DS; ++r) acc2 += hid[r] * sw2[r * DS + t];
    outS[node * DS + t] = s[node * DS + t] + acc2;
}

// ---------------------------------------------------------------------------
extern "C" void kernel_launch(void* const* d_in, const int* in_sizes, int n_in,
                              void* d_out, int out_size, void* d_ws, size_t ws_size,
                              hipStream_t stream) {
    const float* s    = (const float*)d_in[0];
    const float* x    = (const float*)d_in[1];
    const int*   mask = (const int*)  d_in[2];
    const float* ew1  = (const float*)d_in[3];
    const float* eb1  = (const float*)d_in[4];
    const float* ew2  = (const float*)d_in[5];
    const float* eb2  = (const float*)d_in[6];
    const float* aw1  = (const float*)d_in[7];
    const float* ab1  = (const float*)d_in[8];
    const float* aw2  = (const float*)d_in[9];
    const float* ab2  = (const float*)d_in[10];
    const float* sw1  = (const float*)d_in[11];
    const float* sb1  = (const float*)d_in[12];
    const float* sw2  = (const float*)d_in[13];
    const float* sb2  = (const float*)d_in[14];
    const float* cw1  = (const float*)d_in[15];
    const float* cb1  = (const float*)d_in[16];
    const float* cw2  = (const float*)d_in[17];
    const float* cb2  = (const float*)d_in[18];

    float* U  = (float*)d_ws;                 // B*N*32
    float* V  = U + BB * NN * ED;             // B*N*32
    float* MS = V + BB * NN * ED;             // B*N*32

    float* outS = (float*)d_out;              // B*N*128
    float* outX = outS + BB * NN * DS;        // B*N*3

    uv_kernel<<<BB * NN, 64, 0, stream>>>(s, ew1, eb1, U, V);
    edge_kernel<<<BB * NN / 4, 128, 0, stream>>>(x, mask, ew1, ew2, eb2,
                                                 aw1, ab1, aw2, ab2,
                                                 cw1, cb1, cw2, cb2,
                                                 U, V, MS, outX);
    node_kernel<<<BB * NN, 128, 0, stream>>>(s, MS, sw1, sb1, sw2, sb2, outS);
}
